// VariationalAutoencoder_24352464569655
// MI455X (gfx1250) — compile-verified, matrix operand order checked by reading, not run on hardware
//
#include <hip/hip_runtime.h>
#include <stdint.h>

// ---------------------------------------------------------------------------
// Types for CDNA5 WMMA and TDM descriptors
// ---------------------------------------------------------------------------
typedef __attribute__((ext_vector_type(16))) _Float16 v16h;
typedef __attribute__((ext_vector_type(8)))  _Float16 v8h;
typedef __attribute__((ext_vector_type(8)))  float    v8f;
typedef __attribute__((ext_vector_type(4)))  unsigned int u32x4;
typedef __attribute__((ext_vector_type(8)))  unsigned int u32x8;

#define LEAKY(v) ((v) > 0.f ? (v) : 0.01f * (v))

__device__ __forceinline__ float wave_sum(float v) {
#pragma unroll
  for (int off = 16; off > 0; off >>= 1) v += __shfl_down(v, off, 32);
  return v;
}

// ---------------------------------------------------------------------------
// conv1: [2,1,96,96] -> [2,64,48,48], k=4 s=2 p=1, fused LeakyReLU
// ---------------------------------------------------------------------------
__global__ void conv1_leaky(const float* __restrict__ x, const float* __restrict__ w,
                            const float* __restrict__ bias, float* __restrict__ out) {
  int idx = blockIdx.x * blockDim.x + threadIdx.x;      // 2*64*48*48 = 294912
  if (idx >= 294912) return;
  int px = idx % 48, py = (idx / 48) % 48, c = (idx / 2304) % 64, b = idx / (2304 * 64);
  float acc = bias[c];
  const float* xb = x + b * 9216;
  const float* wc = w + c * 16;
#pragma unroll
  for (int ky = 0; ky < 4; ++ky) {
    int iy = py * 2 + ky - 1;
    if (iy < 0 || iy >= 96) continue;
#pragma unroll
    for (int kx = 0; kx < 4; ++kx) {
      int ix = px * 2 + kx - 1;
      if (ix < 0 || ix >= 96) continue;
      acc += xb[iy * 96 + ix] * wc[ky * 4 + kx];
    }
  }
  out[idx] = LEAKY(acc);
}

// ---------------------------------------------------------------------------
// Training-mode BatchNorm over (B,H,W) per channel, in place. B=2 fixed.
// One block per channel.
// ---------------------------------------------------------------------------
__global__ void bn_inplace(float* __restrict__ t, int C, int HW) {
  int c = blockIdx.x;
  int n = 2 * HW;
  float s = 0.f, s2 = 0.f;
  for (int e = threadIdx.x; e < n; e += blockDim.x) {
    int b = (e >= HW); int i = e - b * HW;
    float v = t[((size_t)(b * C + c)) * HW + i];
    s += v; s2 += v * v;
  }
  s = wave_sum(s); s2 = wave_sum(s2);
  __shared__ float r1[8], r2[8];
  int wid = threadIdx.x >> 5, lane = threadIdx.x & 31;
  if (!lane) { r1[wid] = s; r2[wid] = s2; }
  __syncthreads();
  if (threadIdx.x == 0) {
    float S = 0.f, S2 = 0.f;
    for (int i = 0; i < (int)(blockDim.x >> 5); ++i) { S += r1[i]; S2 += r2[i]; }
    float m = S / n;
    float var = S2 / n - m * m;
    r1[0] = m; r2[0] = rsqrtf(var + 1e-5f);
  }
  __syncthreads();
  float m = r1[0], inv = r2[0];
  for (int e = threadIdx.x; e < n; e += blockDim.x) {
    int b = (e >= HW); int i = e - b * HW;
    size_t idx = ((size_t)(b * C + c)) * HW + i;
    t[idx] = (t[idx] - m) * inv;
  }
}

// ---------------------------------------------------------------------------
// fp32 -> fp16 conversion (conv weights)
// ---------------------------------------------------------------------------
__global__ void f32_to_f16(const float* __restrict__ src, _Float16* __restrict__ dst, int n) {
  int i = blockIdx.x * blockDim.x + threadIdx.x;
  if (i < n) dst[i] = (_Float16)src[i];
}

// ---------------------------------------------------------------------------
// im2col, transposed layout Bt[Npad, K], K = Cin*16, f16 output, k=4 s=2 p=1.
// Square images. Pad columns (n >= N) written as zeros.
// ---------------------------------------------------------------------------
__global__ void im2col_f16(const float* __restrict__ in, _Float16* __restrict__ Bt,
                           int Cin, int Hi, int Ho, int N, int Npad) {
  int K = Cin * 16;
  int total = Npad * K;
  for (int idx = blockIdx.x * blockDim.x + threadIdx.x; idx < total;
       idx += gridDim.x * blockDim.x) {
    int n = idx / K, kk = idx % K;
    float v = 0.f;
    if (n < N) {
      int HoWo = Ho * Ho;
      int b = n / HoWo, pix = n % HoWo, y = pix / Ho, x = pix % Ho;
      int ci = kk >> 4, kr = kk & 15, ky = kr >> 2, kx = kr & 3;
      int iy = y * 2 + ky - 1, ix = x * 2 + kx - 1;
      if (iy >= 0 && iy < Hi && ix >= 0 && ix < Hi)
        v = in[(((size_t)b * Cin + ci) * Hi + iy) * Hi + ix];
    }
    Bt[idx] = (_Float16)v;
  }
}

// ---------------------------------------------------------------------------
// WMMA implicit-GEMM conv: C[M,N] = A[M,K] x B[K,N] (f16 in, f32 acc).
// One wave computes a 16x64 output strip (4 accumulators): the A fragment is
// loaded once per K-step and reused by 4 back-to-back v_wmma issues, cutting
// A-operand traffic 4x and raising wmma density in the hot loop.
// A = weights row-major [M,K]; Bt = im2col patches stored [Npad,K] so every
// operand fragment is two contiguous 16B loads matching the 16-bit WMMA
// VGPR layout (lanes 0-15: K b1..b1+7 / b2..b2+7; lanes 16-31: +8).
// Epilogue: + bias, LeakyReLU, scatter to NCHW activations.
// ---------------------------------------------------------------------------
__global__ void wmma_conv_gemm(const _Float16* __restrict__ A, const _Float16* __restrict__ Bt,
                               const float* __restrict__ bias, float* __restrict__ out,
                               int M, int N, int Npad, int K, int HoWo) {
  int wave = (blockIdx.x * blockDim.x + threadIdx.x) >> 5;
  int lane = threadIdx.x & 31;
  int tilesN = Npad >> 6;                  // 64-wide N strips
  int tiles  = (M >> 4) * tilesN;
  if (wave >= tiles) return;
  int tm = wave / tilesN, tn = wave % tilesN;
  int half = lane >> 4, r = lane & 15;

  const _Float16* arow = A + (size_t)(tm * 16 + r) * K;
  const _Float16* brow0 = Bt + (size_t)(tn * 64 + 0  + r) * K;
  const _Float16* brow1 = Bt + (size_t)(tn * 64 + 16 + r) * K;
  const _Float16* brow2 = Bt + (size_t)(tn * 64 + 32 + r) * K;
  const _Float16* brow3 = Bt + (size_t)(tn * 64 + 48 + r) * K;
  const _Float16* brows[4] = {brow0, brow1, brow2, brow3};

  v8f acc[4] = {};
  for (int k0 = 0; k0 < K; k0 += 32) {
    int b1 = k0 + half * 8;       // elements 0..7 : K = b1..b1+7
    int b2 = b1 + 16;             // elements 8..15: K = b2..b2+7
    v8h alo = *(const v8h*)(arow + b1);
    v8h ahi = *(const v8h*)(arow + b2);
    v16h av;
#pragma unroll
    for (int i = 0; i < 8; ++i) { av[i] = alo[i]; av[i + 8] = ahi[i]; }
#pragma unroll
    for (int sub = 0; sub < 4; ++sub) {
      v8h blo = *(const v8h*)(brows[sub] + b1);
      v8h bhi = *(const v8h*)(brows[sub] + b2);
      v16h bv;
#pragma unroll
      for (int i = 0; i < 8; ++i) { bv[i] = blo[i]; bv[i + 8] = bhi[i]; }
      // 8 args: (neg_a, A, neg_b, B, c_mod, C, reuse_a, reuse_b)
      acc[sub] = __builtin_amdgcn_wmma_f32_16x16x32_f16(false, av, false, bv,
                                                        (short)0, acc[sub], false, false);
    }
  }
  // C layout: VGPR i, lanes 0-15 -> M=i, N=lane; lanes 16-31 -> M=8+i, N=lane-16
  int mbase = tm * 16 + half * 8;
#pragma unroll
  for (int sub = 0; sub < 4; ++sub) {
    int n = tn * 64 + sub * 16 + r;
    if (n < N) {
      int b = n / HoWo, pix = n % HoWo;
#pragma unroll
      for (int i = 0; i < 8; ++i) {
        int m = mbase + i;
        float v = acc[sub][i] + bias[m];
        out[((size_t)b * M + m) * HoWo + pix] = LEAKY(v);
      }
    }
  }
}

// ---------------------------------------------------------------------------
// fce GEMV: eigen[b,o] = relu(dot(x4flat[b,:], W[o,:]) + bias[o]),
// K=36864, M=1024, both batches share each weight row (read once).
// One wave per output row, b128 weight streaming + prefetch.
// ---------------------------------------------------------------------------
__global__ void fce_gemv(const float* __restrict__ W, const float* __restrict__ xflat,
                         const float* __restrict__ bias, float* __restrict__ eigen) {
  int wave = (blockIdx.x * blockDim.x + threadIdx.x) >> 5;
  int lane = threadIdx.x & 31;
  if (wave >= 1024) return;
  const float* wr = W + (size_t)wave * 36864;
  const float* xa = xflat;
  const float* xb = xflat + 36864;
  float a0 = 0.f, a1 = 0.f;
  for (int i = lane * 4; i < 36864; i += 128) {
    __builtin_prefetch(wr + i + 2048, 0, 0);
    float4 w4 = *(const float4*)(wr + i);
    float4 p  = *(const float4*)(xa + i);
    float4 q  = *(const float4*)(xb + i);
    a0 += w4.x * p.x + w4.y * p.y + w4.z * p.z + w4.w * p.w;
    a1 += w4.x * q.x + w4.y * q.y + w4.z * q.z + w4.w * q.w;
  }
  a0 = wave_sum(a0); a1 = wave_sum(a1);
  if (!lane) {
    float bb = bias[wave];
    eigen[wave]        = fmaxf(a0 + bb, 0.f);
    eigen[1024 + wave] = fmaxf(a1 + bb, 0.f);
  }
}

// ---------------------------------------------------------------------------
// Spectral helpers (circulant structure of F diag(.) F^H, N=1024):
//   ct[m] = cos(2*pi*m/N);  sin(2*pi*m/N) = ct[(m+768)&1023]
// ---------------------------------------------------------------------------
__global__ void make_costab(float* __restrict__ ct) {
  int i = blockIdx.x * blockDim.x + threadIdx.x;
  if (i < 1024) ct[i] = cosf(6.2831853071795864769f * (float)i * (1.f / 1024.f));
}

// c[b,d] = (1/N) sum_n eigen[b,n] * cos(2*pi*d*n/N)
__global__ void spectral_c(const float* __restrict__ eigen, const float* __restrict__ ct,
                           float* __restrict__ cv) {
  __shared__ float lct[1024];
  for (int i = threadIdx.x; i < 1024; i += blockDim.x) lct[i] = ct[i];
  __syncthreads();
  int wave = blockIdx.x * (blockDim.x >> 5) + (threadIdx.x >> 5);
  int lane = threadIdx.x & 31;
  if (wave >= 2048) return;
  int b = wave >> 10, d = wave & 1023;
  const float* e = eigen + b * 1024;
  float acc = 0.f;
  for (int n = lane; n < 1024; n += 32) acc += e[n] * lct[(d * n) & 1023];
  acc = wave_sum(acc);
  if (!lane) cv[wave] = acc * (1.f / 1024.f);
}

// cov[b,j,k] = c[b, (j-k) mod N]
__global__ void cov_fill(const float* __restrict__ cv, float* __restrict__ cov) {
  int idx = blockIdx.x * blockDim.x + threadIdx.x;      // 2*1024*1024
  if (idx >= 2097152) return;
  int b = idx >> 20;
  int jk = idx & 1048575;
  int j = jk >> 10, k = jk & 1023;
  cov[idx] = cv[(b << 10) + ((j - k) & 1023)];
}

// t[b,n] = (1/sqrt(N)) sum_k exp(+2*pi*i*k*n/N) * (a_k + i b_k)
__global__ void spectral_t(const float* __restrict__ ea, const float* __restrict__ eb,
                           const float* __restrict__ ct,
                           float* __restrict__ tre, float* __restrict__ tim) {
  __shared__ float lct[1024];
  for (int i = threadIdx.x; i < 1024; i += blockDim.x) lct[i] = ct[i];
  __syncthreads();
  int wave = blockIdx.x * (blockDim.x >> 5) + (threadIdx.x >> 5);
  int lane = threadIdx.x & 31;
  if (wave >= 2048) return;
  int b = wave >> 10, n = wave & 1023;
  const float* a = ea + b * 1024;
  const float* q = eb + b * 1024;
  float ar = 0.f, ai = 0.f;
  for (int k = lane; k < 1024; k += 32) {
    int m = (k * n) & 1023;
    float c = lct[m], s = lct[(m + 768) & 1023];
    ar += a[k] * c - q[k] * s;
    ai += a[k] * s + q[k] * c;
  }
  ar = wave_sum(ar); ai = wave_sum(ai);
  if (!lane) { tre[wave] = ar * 0.03125f; tim[wave] = ai * 0.03125f; }
}

// Y[b,j] = Re+Im of (1/sqrt(N)) sum_n exp(-2*pi*i*j*n/N) * e^2[b,n] * t[b,n]
//        = (1/32) sum_n e^2 * ( tr*(c - s) + ti*(c + s) )
__global__ void spectral_y(const float* __restrict__ eigen,
                           const float* __restrict__ tre, const float* __restrict__ tim,
                           const float* __restrict__ ct, float* __restrict__ Y) {
  __shared__ float lct[1024];
  for (int i = threadIdx.x; i < 1024; i += blockDim.x) lct[i] = ct[i];
  __syncthreads();
  int wave = blockIdx.x * (blockDim.x >> 5) + (threadIdx.x >> 5);
  int lane = threadIdx.x & 31;
  if (wave >= 2048) return;
  int b = wave >> 10, j = wave & 1023;
  const float* e  = eigen + b * 1024;
  const float* tr = tre + b * 1024;
  const float* ti = tim + b * 1024;
  float acc = 0.f;
  for (int n = lane; n < 1024; n += 32) {
    int m = (j * n) & 1023;
    float c = lct[m], s = lct[(m + 768) & 1023];
    float ee = e[n] * e[n];
    acc += ee * (tr[n] * (c - s) + ti[n] * (c + s));
  }
  acc = wave_sum(acc);
  if (!lane) Y[wave] = acc * 0.03125f;
}

// ---------------------------------------------------------------------------
// fc1 GEMV: input vector staged into LDS with global_load_async_to_lds_b128
// (per-lane async copy, ASYNCcnt), then each wave streams one weight row.
// ---------------------------------------------------------------------------
__global__ void gemv_fc_async(const float* __restrict__ W, const float* __restrict__ x,
                              const float* __restrict__ bias, float* __restrict__ y,
                              int M, int K) {
  __shared__ float xs[2048];                 // at LDS offset 0 (only shared var)
  int chunks = K >> 2;                       // 16B chunks
  unsigned long long base = (unsigned long long)(uintptr_t)x;
  for (int cidx = threadIdx.x; cidx < chunks; cidx += blockDim.x) {
    unsigned int off = (unsigned int)(cidx * 16);
    // LDS[off] = MEM[base + off], 16 bytes, asynchronous (ASYNCcnt)
    asm volatile("global_load_async_to_lds_b128 %0, %1, %2"
                 :: "v"(off), "v"(off), "s"(base) : "memory");
  }
  asm volatile("s_wait_asynccnt 0" ::: "memory");
  __syncthreads();

  int wave = blockIdx.x * (blockDim.x >> 5) + (threadIdx.x >> 5);
  int lane = threadIdx.x & 31;
  if (wave >= M) return;
  const float* wr = W + (size_t)wave * K;
  float acc = 0.f;
  for (int i = lane * 4; i < K; i += 128) {
    __builtin_prefetch(wr + i + 1024, 0, 0);
    float4 w4 = *(const float4*)(wr + i);
    float4 xv = *(const float4*)(xs + i);
    acc += w4.x * xv.x + w4.y * xv.y + w4.z * xv.z + w4.w * xv.w;
  }
  acc = wave_sum(acc);
  if (!lane) y[wave] = acc + bias[wave];
}

// ---------------------------------------------------------------------------
// fc2 GEMV: input vector staged into LDS by the Tensor Data Mover.
// D# built as a 1-D tile (tile_dim0 = K, data_size = 4B) in 2 SGPR groups,
// issued by wave 0, completion via s_wait_tensorcnt (TENSORcnt).
// ---------------------------------------------------------------------------
__global__ void gemv_fc_tdm(const float* __restrict__ W, const float* __restrict__ x,
                            const float* __restrict__ bias, float* __restrict__ y,
                            int M, int K) {
  __shared__ float xs[8704];                 // at LDS offset 0 (only shared var)
  if ((threadIdx.x >> 5) == 0) {
    unsigned long long ga = (unsigned long long)(uintptr_t)x;
    unsigned int uK = (unsigned int)K;
    u32x4 g0;
    g0[0] = 1u;                                            // count=1, user descriptor
    g0[1] = 0u;                                            // lds_addr = 0
    g0[2] = (unsigned int)(ga & 0xFFFFFFFFu);              // global_addr[31:0]
    g0[3] = (unsigned int)((ga >> 32) & 0x01FFFFFFu)       // global_addr[56:32]
            | 0x80000000u;                                 // type = 2 ("image")
    u32x8 g1;
    g1[0] = 2u << 16;                                      // data_size = 4B
    g1[1] = (uK & 0xFFFFu) << 16;                          // tensor_dim0[15:0]
    g1[2] = ((uK >> 16) & 0xFFFFu) | (1u << 16);           // tensor_dim0[31:16], tensor_dim1=1
    g1[3] = (uK & 0xFFFFu) << 16;                          // tile_dim0 = K (fits 16b)
    g1[4] = 1u;                                            // tile_dim1 = 1
    g1[5] = uK;                                            // tensor_dim0_stride[31:0]
    g1[6] = 0u;
    g1[7] = 0u;
    asm volatile("tensor_load_to_lds %0, %1" :: "s"(g0), "s"(g1) : "memory");
    __builtin_amdgcn_s_wait_tensorcnt(0);
  }
  __syncthreads();

  int wave = blockIdx.x * (blockDim.x >> 5) + (threadIdx.x >> 5);
  int lane = threadIdx.x & 31;
  if (wave >= M) return;
  const float* wr = W + (size_t)wave * K;
  float acc = 0.f;
  for (int i = lane * 4; i < K; i += 128) {
    __builtin_prefetch(wr + i + 1024, 0, 0);
    float4 w4 = *(const float4*)(wr + i);
    float4 xv = *(const float4*)(xs + i);
    acc += w4.x * xv.x + w4.y * xv.y + w4.z * xv.z + w4.w * xv.w;
  }
  acc = wave_sum(acc);
  if (!lane) y[wave] = acc + bias[wave];
}

// ---------------------------------------------------------------------------
// ConvTranspose2d (padding=0, k=5, square), gather form:
//   out[b,co,y,x] = bias[co] + sum in[b,ci,iy,ix]*w[ci,co,ky,kx], y=iy*s+ky
// act: 0 = LeakyReLU(0.01), 1 = sigmoid. B=2 fixed.
// ---------------------------------------------------------------------------
__global__ void convT_k(const float* __restrict__ in, const float* __restrict__ w,
                        const float* __restrict__ bias, float* __restrict__ out,
                        int Cin, int Cout, int Hin, int Hout, int stride, int act) {
  int total = 2 * Cout * Hout * Hout;
  int idx = blockIdx.x * blockDim.x + threadIdx.x;
  if (idx >= total) return;
  int x = idx % Hout, y = (idx / Hout) % Hout;
  int co = (idx / (Hout * Hout)) % Cout, b = idx / (Hout * Hout * Cout);
  float acc = bias[co];
  const float* ib = in + (size_t)b * Cin * Hin * Hin;
#pragma unroll
  for (int ky = 0; ky < 5; ++ky) {
    int ty = y - ky;
    if (ty < 0 || (ty % stride)) continue;
    int iy = ty / stride;
    if (iy >= Hin) continue;
#pragma unroll
    for (int kx = 0; kx < 5; ++kx) {
      int tx = x - kx;
      if (tx < 0 || (tx % stride)) continue;
      int ix = tx / stride;
      if (ix >= Hin) continue;
      const float* ip = ib + (size_t)iy * Hin + ix;
      const float* wp = w + (size_t)co * 25 + ky * 5 + kx;
      float a = 0.f;
      for (int ci = 0; ci < Cin; ++ci)
        a += ip[(size_t)ci * Hin * Hin] * wp[(size_t)ci * Cout * 25];
      acc += a;
    }
  }
  out[idx] = (act == 0) ? LEAKY(acc) : (1.f / (1.f + __expf(-acc)));
}

// ---------------------------------------------------------------------------
// Host orchestration
// ---------------------------------------------------------------------------
extern "C" void kernel_launch(void* const* d_in, const int* in_sizes, int n_in,
                              void* d_out, int out_size, void* d_ws, size_t ws_size,
                              hipStream_t stream) {
  (void)in_sizes; (void)n_in; (void)out_size; (void)ws_size;
  const float* x     = (const float*)d_in[0];
  const float* eps_a = (const float*)d_in[1];
  const float* eps_b = (const float*)d_in[2];
  const float* ec1_w = (const float*)d_in[3];  const float* ec1_b = (const float*)d_in[4];
  const float* ec2_w = (const float*)d_in[5];  const float* ec2_b = (const float*)d_in[6];
  const float* ec3_w = (const float*)d_in[7];  const float* ec3_b = (const float*)d_in[8];
  const float* ec4_w = (const float*)d_in[9];  const float* ec4_b = (const float*)d_in[10];
  const float* fce_w = (const float*)d_in[11]; const float* fce_b = (const float*)d_in[12];
  const float* fc1_w = (const float*)d_in[13]; const float* fc1_b = (const float*)d_in[14];
  const float* fc2_w = (const float*)d_in[15]; const float* fc2_b = (const float*)d_in[16];
  const float* dc1_w = (const float*)d_in[17]; const float* dc1_b = (const float*)d_in[18];
  const float* dc2_w = (const float*)d_in[19]; const float* dc2_b = (const float*)d_in[20];
  const float* dc3_w = (const float*)d_in[21]; const float* dc3_b = (const float*)d_in[22];
  const float* dc4_w = (const float*)d_in[23]; const float* dc4_b = (const float*)d_in[24];
  const float* dc5_w = (const float*)d_in[25]; const float* dc5_b = (const float*)d_in[26];
  const float* dc6_w = (const float*)d_in[27]; const float* dc6_b = (const float*)d_in[28];

  // bump allocator in d_ws (256B aligned regions)
  size_t off = 0;
  auto alloc = [&](size_t bytes) -> char* {
    size_t o = (off + 255) & ~(size_t)255;
    off = o + bytes;
    return (char*)d_ws + o;
  };
  float*    a1    = (float*)alloc(294912u * 4);   // [2,64,48,48]
  float*    a2    = (float*)alloc(294912u * 4);   // [2,256,24,24]
  float*    a3    = (float*)alloc(147456u * 4);   // [2,512,12,12]
  float*    a4    = (float*)alloc(73728u  * 4);   // [2,1024,6,6]
  _Float16* wh    = (_Float16*)alloc(8388608u * 2); // largest f16 weight (conv4)
  _Float16* bt    = (_Float16*)alloc(1310720u * 2); // largest im2col buffer (conv3: 320*4096)
  float*    eig   = (float*)alloc(2048u * 4);
  float*    ct    = (float*)alloc(1024u * 4);
  float*    cv    = (float*)alloc(2048u * 4);
  float*    tre   = (float*)alloc(2048u * 4);
  float*    tim   = (float*)alloc(2048u * 4);
  float*    Yv    = (float*)alloc(2048u * 4);
  float*    h1    = (float*)alloc(8704u  * 4);
  float*    h2    = (float*)alloc(16384u * 4);
  float*    dbuf1 = (float*)alloc(46208u  * 4);   // [2,64,19,19]
  float*    dbuf2 = (float*)alloc(107584u * 4);   // [2,32,41,41]
  float*    dbuf3 = (float*)alloc(231200u * 4);   // [2,16,85,85]
  float*    dbuf4 = (float*)alloc(126736u * 4);   // [2,8,89,89]
  float*    dbuf5 = (float*)alloc(69192u  * 4);   // [2,4,93,93]

  float* x_recon = (float*)d_out;                 // [2,1,97,97] = 18818
  float* cov     = (float*)d_out + 18818;         // [2,1024,1024]

  auto cdiv = [](int a, int b) { return (a + b - 1) / b; };

  // ---- Encoder ----
  conv1_leaky<<<1152, 256, 0, stream>>>(x, ec1_w, ec1_b, a1);
  bn_inplace<<<64, 256, 0, stream>>>(a1, 64, 2304);

  // conv2: M=256 N=1152 (Npad=1152, 18 strips) K=1024 -> 16*18=288 waves
  f32_to_f16<<<cdiv(262144, 256), 256, 0, stream>>>(ec2_w, wh, 262144);
  im2col_f16<<<cdiv(1179648, 256), 256, 0, stream>>>(a1, bt, 64, 48, 24, 1152, 1152);
  wmma_conv_gemm<<<cdiv(288 * 32, 256), 256, 0, stream>>>(wh, bt, ec2_b, a2,
                                                          256, 1152, 1152, 1024, 576);
  bn_inplace<<<256, 256, 0, stream>>>(a2, 256, 576);

  // conv3: M=512 N=288 (Npad=320, 5 strips) K=4096 -> 32*5=160 waves
  f32_to_f16<<<cdiv(2097152, 256), 256, 0, stream>>>(ec3_w, wh, 2097152);
  im2col_f16<<<cdiv(1310720, 256), 256, 0, stream>>>(a2, bt, 256, 24, 12, 288, 320);
  wmma_conv_gemm<<<cdiv(160 * 32, 256), 256, 0, stream>>>(wh, bt, ec3_b, a3,
                                                          512, 288, 320, 4096, 144);
  bn_inplace<<<512, 256, 0, stream>>>(a3, 512, 144);

  // conv4: M=1024 N=72 (Npad=128, 2 strips) K=8192 -> 64*2=128 waves
  f32_to_f16<<<cdiv(8388608, 256), 256, 0, stream>>>(ec4_w, wh, 8388608);
  im2col_f16<<<cdiv(1048576, 256), 256, 0, stream>>>(a3, bt, 512, 12, 6, 72, 128);
  wmma_conv_gemm<<<cdiv(128 * 32, 256), 256, 0, stream>>>(wh, bt, ec4_b, a4,
                                                          1024, 72, 128, 8192, 36);
  bn_inplace<<<1024, 256, 0, stream>>>(a4, 1024, 36);

  fce_gemv<<<128, 256, 0, stream>>>(fce_w, a4, fce_b, eig);

  // ---- Latent sample (circulant structure of F diag(.) F^H) ----
  make_costab<<<4, 256, 0, stream>>>(ct);
  spectral_c<<<256, 256, 0, stream>>>(eig, ct, cv);
  cov_fill<<<8192, 256, 0, stream>>>(cv, cov);
  spectral_t<<<256, 256, 0, stream>>>(eps_a, eps_b, ct, tre, tim);
  spectral_y<<<256, 256, 0, stream>>>(eig, tre, tim, ct, Yv);

  // ---- Decoder FC (memory-bound GEMVs; LDS-resident input vector) ----
  gemv_fc_async<<<1088, 256, 0, stream>>>(fc1_w, Yv, fc1_b, h1, 8704, 2048);
  gemv_fc_tdm<<<2048, 256, 0, stream>>>(fc2_w, h1, fc2_b, h2, 16384, 8704);

  // ---- Decoder ConvT chain ----
  convT_k<<<cdiv(46208, 256),  256, 0, stream>>>(h2,    dc1_w, dc1_b, dbuf1, 128, 64, 8,  19, 2, 0);
  convT_k<<<cdiv(107584, 256), 256, 0, stream>>>(dbuf1, dc2_w, dc2_b, dbuf2, 64,  32, 19, 41, 2, 0);
  convT_k<<<cdiv(231200, 256), 256, 0, stream>>>(dbuf2, dc3_w, dc3_b, dbuf3, 32,  16, 41, 85, 2, 0);
  convT_k<<<cdiv(126736, 256), 256, 0, stream>>>(dbuf3, dc4_w, dc4_b, dbuf4, 16,  8,  85, 89, 1, 0);
  convT_k<<<cdiv(69192, 256),  256, 0, stream>>>(dbuf4, dc5_w, dc5_b, dbuf5, 8,   4,  89, 93, 1, 0);
  convT_k<<<cdiv(18818, 256),  256, 0, stream>>>(dbuf5, dc6_w, dc6_b, x_recon, 4, 1,  93, 97, 1, 1);
}
